// SS2Dv2_4973572129178
// MI455X (gfx1250) — matrix operand order repaired and checked by reading, not compiled, not run on hardware
//
#include <hip/hip_runtime.h>
#include <hip/hip_bf16.h>

typedef __attribute__((ext_vector_type(16))) _Float16 v16h;
typedef __attribute__((ext_vector_type(8)))  float    v8f;

// Problem constants
#define DM_   96
#define DI_   192
#define NS_   16
#define KK_   4
#define RR_   6
#define HH_   56
#define WW_   56
#define LL_   3136
#define NCH_  56   // number of scan chunks
#define CHL_  56   // chunk length (NCH_*CHL_ == LL_)

// ---------------------------------------------------------------------------
// WMMA NT tile with split-f16 (hi+lo) operands for ~f32 accuracy:
//   C(16x16) += A[m,:] * B[n,:]^T,  K multiple of 32, f32 accumulate.
// Per K-step: 3x v_wmma_f32_16x16x32_f16 (hi*hi, lo*hi, hi*lo).
// A fragment layout (16x32 f16): lanes 0-15 -> M=lane, halves 0-7 = K0..7,
// halves 8-15 = K16..23; lanes 16-31 -> M=lane-16, K8..15 / K24..31.
// B fragment layout (32x16 f16): same K order as A; lanes 0-15 -> N=lane,
// K0..7 / K16..23; lanes 16-31 -> N=lane-16, K8..15 / K24..31.
// C/D layout: VGPR r, lanes 0-15 -> (M=r, N=lane); lanes 16-31 -> (M=r+8).
// ---------------------------------------------------------------------------
template <int KDIM>
__device__ inline v8f wmma_nt_tile(const float* __restrict__ A, int lda, int mbase, int Mv,
                                   const float* __restrict__ B, int ldb, int nbase,
                                   int lane) {
  v8f acc = {};
  const int hs   = lane >> 4;          // half-wave select (0/1)
  const int mrow = mbase + (lane & 15);
  const int ncol = nbase + (lane & 15);
  const bool mok = (mrow < Mv);
  const float* arow = A + (mok ? mrow : 0) * lda;   // clamped: safe to read
  const float* brow = B + ncol * ldb;
#pragma unroll
  for (int k0 = 0; k0 < KDIM; k0 += 32) {
    v16h ah, al, bh, bl;
    const int ka0 = k0 + hs * 8;       // first  8 K's for this lane group
    const int ka1 = ka0 + 16;          // second 8 K's
#pragma unroll
    for (int h = 0; h < 8; ++h) {
      float v0 = mok ? arow[ka0 + h] : 0.f;
      float v1 = mok ? arow[ka1 + h] : 0.f;
      _Float16 h0 = (_Float16)v0;
      _Float16 h1 = (_Float16)v1;
      ah[h]     = h0;  al[h]     = (_Float16)(v0 - (float)h0);
      ah[h + 8] = h1;  al[h + 8] = (_Float16)(v1 - (float)h1);
    }
    const int kb = k0 + hs * 8;
#pragma unroll
    for (int h = 0; h < 16; ++h) {
      float v = brow[kb + h + (h >= 8 ? 8 : 0)];
      _Float16 h0 = (_Float16)v;
      bh[h] = h0;  bl[h] = (_Float16)(v - (float)h0);
    }
    acc = __builtin_amdgcn_wmma_f32_16x16x32_f16(false, ah, false, bh, (short)0, acc, false, false);
    acc = __builtin_amdgcn_wmma_f32_16x16x32_f16(false, al, false, bh, (short)0, acc, false, false);
    acc = __builtin_amdgcn_wmma_f32_16x16x32_f16(false, ah, false, bl, (short)0, acc, false, false);
  }
  return acc;
}

// ---------------------------------------------------------------------------
// GEMM1: xz[l][o] = sum_c x[l][c] * in_proj_w[o][c]   (3136x384, K=96)
// ---------------------------------------------------------------------------
__global__ void k_gemm_inproj(const float* __restrict__ x,
                              const float* __restrict__ w,
                              float* __restrict__ xz) {
  const int lane = threadIdx.x & 31;
  const int tile = blockIdx.x * (blockDim.x >> 5) + (threadIdx.x >> 5);
  const int NT = 2 * DI_ / 16;                 // 24
  if (tile >= (LL_ / 16) * NT) return;
  const int mt = tile / NT, nt = tile % NT;
  v8f acc = wmma_nt_tile<DM_>(x, DM_, mt * 16, LL_, w, DM_, nt * 16, lane);
  const int n = nt * 16 + (lane & 15);
#pragma unroll
  for (int r = 0; r < 8; ++r) {
    const int m = mt * 16 + r + ((lane >> 4) << 3);
    xz[m * (2 * DI_) + n] = acc[r];
  }
}

// ---------------------------------------------------------------------------
// Depthwise 3x3 conv (SAME) + bias + SiLU, fused with cross-scan gather:
// writes xs_t[k][j][d] for the 4 scan directions.
// ---------------------------------------------------------------------------
__global__ void k_conv_scan(const float* __restrict__ xz,
                            const float* __restrict__ cw,
                            const float* __restrict__ cb,
                            float* __restrict__ xs_t) {
  const int l = blockIdx.x, d = threadIdx.x;   // 192 threads
  const int h = l / WW_, w = l % WW_;
  float acc = cb[d];
#pragma unroll
  for (int dh = -1; dh <= 1; ++dh)
#pragma unroll
    for (int dw = -1; dw <= 1; ++dw) {
      const int hh = h + dh, ww = w + dw;
      if (hh >= 0 && hh < HH_ && ww >= 0 && ww < WW_)
        acc += xz[(hh * WW_ + ww) * (2 * DI_) + d] * cw[d * 9 + (dh + 1) * 3 + (dw + 1)];
    }
  const float xv = acc / (1.f + __expf(-acc));   // SiLU
  const int j1 = w * HH_ + h;                    // transposed traversal index
  xs_t[(0 * LL_ + l) * DI_ + d] = xv;
  xs_t[(1 * LL_ + j1) * DI_ + d] = xv;
  xs_t[(2 * LL_ + (LL_ - 1 - l)) * DI_ + d] = xv;
  xs_t[(3 * LL_ + (LL_ - 1 - j1)) * DI_ + d] = xv;
}

// ---------------------------------------------------------------------------
// GEMM2: x_dbl[k][r][j] = sum_d x_proj_w[k][r][d] * xs_t[k][j][d]
// rows 0..5 -> dts[k][r][j]; rows 6..37 -> bc[k][j][0..31] (B then C).
// M=38 padded to 48 (3 tiles), masked.
// ---------------------------------------------------------------------------
__global__ void k_gemm_xproj(const float* __restrict__ xpw,
                             const float* __restrict__ xs_t,
                             float* __restrict__ dts,
                             float* __restrict__ bc) {
  const int lane = threadIdx.x & 31;
  const int tile = blockIdx.x * (blockDim.x >> 5) + (threadIdx.x >> 5);
  const int TPK = 3 * (LL_ / 16);              // tiles per k = 588
  if (tile >= KK_ * TPK) return;
  const int k = tile / TPK, rem = tile % TPK;
  const int mt = rem / (LL_ / 16), nt = rem % (LL_ / 16);
  const float* A = xpw + k * (RR_ + 2 * NS_) * DI_;
  const float* B = xs_t + (size_t)k * LL_ * DI_;
  v8f acc = wmma_nt_tile<DI_>(A, DI_, mt * 16, RR_ + 2 * NS_, B, DI_, nt * 16, lane);
  const int j = nt * 16 + (lane & 15);
#pragma unroll
  for (int r = 0; r < 8; ++r) {
    const int rr = mt * 16 + r + ((lane >> 4) << 3);
    const float v = acc[r];
    if (rr < RR_)                 dts[(k * RR_ + rr) * LL_ + j] = v;
    else if (rr < RR_ + 2 * NS_)  bc[((size_t)k * LL_ + j) * 32 + (rr - RR_)] = v;
  }
}

// ---------------------------------------------------------------------------
// delta[k][d][j] = softplus( sum_r dt_projs_w[k][d][r]*dts[k][r][j] + bias )
// ---------------------------------------------------------------------------
__global__ void k_dt(const float* __restrict__ dts,
                     const float* __restrict__ dtw,
                     const float* __restrict__ dtb,
                     float* __restrict__ delta) {
  const int kd = blockIdx.x;
  const int k = kd / DI_, d = kd % DI_;
  float wr[RR_];
#pragma unroll
  for (int r = 0; r < RR_; ++r) wr[r] = dtw[(k * DI_ + d) * RR_ + r];
  const float bias = dtb[kd];
  for (int j = threadIdx.x; j < LL_; j += blockDim.x) {
    float s = bias;
#pragma unroll
    for (int r = 0; r < RR_; ++r) s += wr[r] * dts[(k * RR_ + r) * LL_ + j];
    delta[(size_t)kd * LL_ + j] = (s > 20.f) ? s : log1pf(__expf(s));
  }
}

// ---------------------------------------------------------------------------
// Chunked scan, pass 1: per (k,d,chunk), lanes 0..15 carry state n.
// local recurrence h = exp(delta*A_n)*h + delta*B_n*u ; also product of dA.
// ---------------------------------------------------------------------------
__global__ void k_scan_pass1(const float* __restrict__ delta,
                             const float* __restrict__ xs_t,
                             const float* __restrict__ bc,
                             const float* __restrict__ A_logs,
                             float* __restrict__ hsum,
                             float* __restrict__ aprodb) {
  const int unit = blockIdx.x * (blockDim.x >> 4) + (threadIdx.x >> 4);
  const int n = threadIdx.x & 15;
  const int c = unit % NCH_;
  const int kd = unit / NCH_;
  const int d = kd % DI_, k = kd / DI_;
  const float an = -__expf(A_logs[kd * NS_ + n]);
  const float* dptr = delta + (size_t)kd * LL_ + c * CHL_;
  const float* uptr = xs_t + ((size_t)k * LL_ + c * CHL_) * DI_ + d;
  const float* bptr = bc + ((size_t)k * LL_ + c * CHL_) * 32 + n;
  float h = 0.f, ap = 1.f;
  for (int i = 0; i < CHL_; ++i) {
    const float dv = dptr[i];
    const float u  = uptr[(size_t)i * DI_];
    const float bn = bptr[(size_t)i * 32];
    const float e = __expf(dv * an);
    h = e * h + dv * bn * u;
    ap *= e;
  }
  const size_t idx = ((size_t)kd * NCH_ + c) * NS_ + n;
  hsum[idx] = h;
  aprodb[idx] = ap;
}

// Pass 2: serial prefix over the 56 chunk summaries per (k,d) -> incoming h.
__global__ void k_scan_pass2(const float* __restrict__ hsum,
                             const float* __restrict__ aprodb,
                             float* __restrict__ hin) {
  const int kd = blockIdx.x * (blockDim.x >> 4) + (threadIdx.x >> 4);
  const int n = threadIdx.x & 15;
  float carry = 0.f;
  for (int c = 0; c < NCH_; ++c) {
    const size_t idx = ((size_t)kd * NCH_ + c) * NS_ + n;
    hin[idx] = carry;
    carry = aprodb[idx] * carry + hsum[idx];
  }
}

// Pass 3: rerun chunk with incoming carry, project y = sum_n h*C_n + D*u.
__global__ void k_scan_pass3(const float* __restrict__ delta,
                             const float* __restrict__ xs_t,
                             const float* __restrict__ bc,
                             const float* __restrict__ A_logs,
                             const float* __restrict__ Ds,
                             const float* __restrict__ hin,
                             float* __restrict__ ybuf) {
  const int unit = blockIdx.x * (blockDim.x >> 4) + (threadIdx.x >> 4);
  const int n = threadIdx.x & 15;
  const int c = unit % NCH_;
  const int kd = unit / NCH_;
  const int d = kd % DI_, k = kd / DI_;
  const float an = -__expf(A_logs[kd * NS_ + n]);
  const float Dv = Ds[kd];
  float h = hin[((size_t)kd * NCH_ + c) * NS_ + n];
  const float* dptr = delta + (size_t)kd * LL_ + c * CHL_;
  const float* uptr = xs_t + ((size_t)k * LL_ + c * CHL_) * DI_ + d;
  const float* bptr = bc + ((size_t)k * LL_ + c * CHL_) * 32 + n;
  for (int i = 0; i < CHL_; ++i) {
    const float dv = dptr[i];
    const float u  = uptr[(size_t)i * DI_];
    const float bn = bptr[(size_t)i * 32];
    const float cn = bptr[(size_t)i * 32 + 16];
    const float e = __expf(dv * an);
    h = e * h + dv * bn * u;
    float v = h * cn;
    v += __shfl_xor(v, 1, 16);
    v += __shfl_xor(v, 2, 16);
    v += __shfl_xor(v, 4, 16);
    v += __shfl_xor(v, 8, 16);
    if (n == 0) ybuf[((size_t)k * LL_ + c * CHL_ + i) * DI_ + d] = v + Dv * u;
  }
}

// ---------------------------------------------------------------------------
// Cross-merge (4 directions) + LayerNorm over DI + SiLU(z) gate.
// ---------------------------------------------------------------------------
__global__ void k_merge_ln(const float* __restrict__ ybuf,
                           const float* __restrict__ xz,
                           const float* __restrict__ lnw,
                           const float* __restrict__ lnb,
                           float* __restrict__ yln) {
  __shared__ float sred[8];
  const int l = blockIdx.x, d = threadIdx.x;   // 192 threads = 6 waves
  const int h = l / WW_, w = l % WW_;
  const int j1 = w * HH_ + h;
  float y = ybuf[((size_t)0 * LL_ + l) * DI_ + d]
          + ybuf[((size_t)1 * LL_ + j1) * DI_ + d]
          + ybuf[((size_t)2 * LL_ + (LL_ - 1 - l)) * DI_ + d]
          + ybuf[((size_t)3 * LL_ + (LL_ - 1 - j1)) * DI_ + d];
  // mean over DI
  float s = y;
#pragma unroll
  for (int m = 16; m >= 1; m >>= 1) s += __shfl_xor(s, m, 32);
  if ((threadIdx.x & 31) == 0) sred[threadIdx.x >> 5] = s;
  __syncthreads();
  const float mu = (sred[0] + sred[1] + sred[2] + sred[3] + sred[4] + sred[5]) / (float)DI_;
  __syncthreads();
  const float dv = y - mu;
  float s2 = dv * dv;
#pragma unroll
  for (int m = 16; m >= 1; m >>= 1) s2 += __shfl_xor(s2, m, 32);
  if ((threadIdx.x & 31) == 0) sred[threadIdx.x >> 5] = s2;
  __syncthreads();
  const float var = (sred[0] + sred[1] + sred[2] + sred[3] + sred[4] + sred[5]) / (float)DI_;
  const float zq = xz[(size_t)l * (2 * DI_) + DI_ + d];
  const float zs = zq / (1.f + __expf(-zq));
  yln[(size_t)l * DI_ + d] = (dv * rsqrtf(var + 1e-5f) * lnw[d] + lnb[d]) * zs;
}

// ---------------------------------------------------------------------------
// GEMM4: out[l][o] = sum_d yln[l][d] * out_proj_w[o][d]   (3136x96, K=192)
// ---------------------------------------------------------------------------
__global__ void k_gemm_outproj(const float* __restrict__ yln,
                               const float* __restrict__ w,
                               float* __restrict__ out) {
  const int lane = threadIdx.x & 31;
  const int tile = blockIdx.x * (blockDim.x >> 5) + (threadIdx.x >> 5);
  const int NT = DM_ / 16;                     // 6
  if (tile >= (LL_ / 16) * NT) return;
  const int mt = tile / NT, nt = tile % NT;
  v8f acc = wmma_nt_tile<DI_>(yln, DI_, mt * 16, LL_, w, DI_, nt * 16, lane);
  const int o = nt * 16 + (lane & 15);
#pragma unroll
  for (int r = 0; r < 8; ++r) {
    const int m = mt * 16 + r + ((lane >> 4) << 3);
    out[(size_t)m * DM_ + o] = acc[r];
  }
}

// ---------------------------------------------------------------------------
extern "C" void kernel_launch(void* const* d_in, const int* in_sizes, int n_in,
                              void* d_out, int out_size, void* d_ws, size_t ws_size,
                              hipStream_t stream) {
  const float* x            = (const float*)d_in[0];
  const float* in_proj_w    = (const float*)d_in[1];
  const float* conv_w       = (const float*)d_in[2];
  const float* conv_b       = (const float*)d_in[3];
  const float* x_proj_w     = (const float*)d_in[4];
  const float* dt_projs_w   = (const float*)d_in[5];
  const float* dt_projs_b   = (const float*)d_in[6];
  const float* A_logs       = (const float*)d_in[7];
  const float* Ds           = (const float*)d_in[8];
  const float* ln_w         = (const float*)d_in[9];
  const float* ln_b         = (const float*)d_in[10];
  const float* out_proj_w   = (const float*)d_in[11];
  float* out = (float*)d_out;

  // Workspace layout (float offsets); total ~46.3 MB
  float* ws    = (float*)d_ws;
  float* xz    = ws;                 // [L][384]            1,204,224
  float* xs_t  = ws + 1204224;       // [K][L][DI]          2,408,448
  float* dts   = ws + 3612672;       // [K][R][L]              75,264
  float* bcbuf = ws + 3687936;       // [K][L][32]            401,408
  float* delta = ws + 4089344;       // [K*DI][L]           2,408,448
  float* hsum  = ws + 6497792;       // [K*DI][NCH][N]        688,128
  float* aprod = ws + 7185920;       //                       688,128
  float* hin   = ws + 7874048;       //                       688,128
  float* ybuf  = ws + 8562176;       // [K][L][DI]          2,408,448
  float* yln   = ws + 10970624;      // [L][DI]               602,112

  k_gemm_inproj<<<588, 256, 0, stream>>>(x, in_proj_w, xz);                 // 4704 tiles
  k_conv_scan  <<<LL_, DI_, 0, stream>>>(xz, conv_w, conv_b, xs_t);
  k_gemm_xproj <<<294, 256, 0, stream>>>(x_proj_w, xs_t, dts, bcbuf);       // 2352 tiles
  k_dt         <<<KK_ * DI_, 256, 0, stream>>>(dts, dt_projs_w, dt_projs_b, delta);
  k_scan_pass1 <<<2688, 256, 0, stream>>>(delta, xs_t, bcbuf, A_logs, hsum, aprod);
  k_scan_pass2 <<<48, 256, 0, stream>>>(hsum, aprod, hin);
  k_scan_pass3 <<<2688, 256, 0, stream>>>(delta, xs_t, bcbuf, A_logs, Ds, hin, ybuf);
  k_merge_ln   <<<LL_, DI_, 0, stream>>>(ybuf, xz, ln_w, ln_b, yln);
  k_gemm_outproj<<<147, 256, 0, stream>>>(yln, out_proj_w, out);            // 1176 tiles
}
